// DecoderRNN_56032143344183
// MI455X (gfx1250) — matrix operand order repaired and checked by reading, not compiled, not run on hardware
//
#include <hip/hip_runtime.h>
#include <hip/hip_bf16.h>
#include <math.h>

// Problem constants (from reference)
#define Bn 128
#define Pn 196
#define En 256
#define An 256
#define Dn 512
#define Mn 512
#define Vn 10000
#define Ln 26
#define Tn 25   // L - 1

typedef __attribute__((ext_vector_type(16))) _Float16 v16h;
typedef __attribute__((ext_vector_type(8)))  float    v8f;
typedef __attribute__((ext_vector_type(4)))  float    f32x4;
typedef __attribute__((ext_vector_type(4)))  _Float16 h16x4;

// ---------------------------------------------------------------------------
// WMMA GEMM:  C[M x N] = A[M x K] (f32, row-major) @ Bw[N x K]^T (f16) + bias
//
// Block tile 64(M) x 64(N), K-chunk 32, double-buffered LDS, software
// pipelined. A tiles (f32, need conversion) go global->VGPR->cvt->LDS.
// B tiles (already f16) use CDNA5 async copies: GLOBAL_LOAD_ASYNC_TO_LDS_B64
// (no VGPR staging, tracked by ASYNCcnt, overlaps the WMMAs).
// 256 threads = 8 wave32s; each wave owns one 16-row strip and two 16x16
// N-tiles -> 2 x v_wmma_f32_16x16x32_f16 per wave per K-chunk.
//
// REQUIREMENTS: M % 64 == 0, K % 32 == 0 (true for all call sites).
// N may be ragged (pred N=10000): B-row loads are clamped, stores guarded.
// MASKED=1: zero rows where rowmask[m]==0 (masked predictions).
// ---------------------------------------------------------------------------
#define LDP 40   // LDS row stride in halves (80 B) -> keeps b128 frag reads aligned

template <int MASKED>
__global__ __launch_bounds__(256)
void gemm_wmma_f16(const float* __restrict__ A,
                   const _Float16* __restrict__ Bw,
                   const float* __restrict__ bias,
                   float* __restrict__ C, long ldc,
                   int N, int K,
                   const int* __restrict__ rowmask)
{
    __shared__ __align__(16) _Float16 As[2][64 * LDP];
    __shared__ __align__(16) _Float16 Bs[2][64 * LDP];

    const int tid  = threadIdx.x;
    const int m0   = blockIdx.y * 64;
    const int n0   = blockIdx.x * 64;
    const int w    = tid >> 5;
    const int lane = tid & 31;
    const int wm   = w & 3;          // 4 M-tiles of 16
    const int wn   = w >> 2;         // 2 N-pairs (cols wn*16 and wn*16+32)
    const int hi   = lane >> 4;
    const int lr   = lane & 15;

    // staging coordinates: each thread covers rows {tid/8, tid/8+32}, 4 halves
    const int srow = tid >> 3;
    const int sk   = (tid & 7) * 4;

    v8f acc0 = {}, acc1 = {};
    const int nk = K >> 5;

    f32x4 aReg[2];

    // A tile: f32 global -> regs (for f16 convert)
    auto gloadA = [&](int kc) {
#pragma unroll
        for (int j = 0; j < 2; ++j) {
            int row = srow + j * 32;
            aReg[j] = *(const f32x4*)(A + (long)(m0 + row) * K + kc * 32 + sk);
        }
    };
    auto sstoreA = [&](int buf) {
#pragma unroll
        for (int j = 0; j < 2; ++j) {
            int row = srow + j * 32;
            h16x4 hv;
            hv[0] = (_Float16)aReg[j][0];
            hv[1] = (_Float16)aReg[j][1];
            hv[2] = (_Float16)aReg[j][2];
            hv[3] = (_Float16)aReg[j][3];
            *(h16x4*)&As[buf][row * LDP + sk] = hv;
        }
    };
    // B tile: f16 weights, async copy global -> LDS (no VGPR staging).
    // FLAT/async LDS addressing: generic LDS address truncated to 32 bits is
    // the LDS byte offset (addr[31:0]).
    auto bAsync = [&](int kc, int buf) {
#pragma unroll
        for (int j = 0; j < 2; ++j) {
            int row = srow + j * 32;
            int gn  = n0 + row;
            gn = (gn < N) ? gn : (N - 1);       // clamp, stores guarded later
            unsigned long long ga =
                (unsigned long long)(size_t)(Bw + (long)gn * K + kc * 32 + sk);
            unsigned ldsOff = (unsigned)(size_t)(&Bs[buf][row * LDP + sk]);
            asm volatile("global_load_async_to_lds_b64 %0, %1, off"
                         :: "v"(ldsOff), "v"(ga) : "memory");
        }
    };
    auto compute = [&](int buf) {
        // A fragment (16-bit 16x32 layout): lanes 0-15 K={0..7,16..23},
        // lanes 16-31 K={8..15,24..31}
        v16h af, bf0, bf1;
        const int am  = wm * 16 + lr;
        const int bnA = wn * 16 + lr;
        const int bnB = bnA + 32;
#pragma unroll
        for (int e = 0; e < 16; ++e) {
            int ka = e + ((e >= 8) ? 8 : 0) + (hi ? 8 : 0);
            af[e] = As[buf][am * LDP + ka];
        }
#pragma unroll
        for (int e = 0; e < 16; ++e) {
            int kb = e + ((e >= 8) ? 8 : 0) + (hi ? 8 : 0);       // B: lane->N col, halves->K
            bf0[e] = Bs[buf][bnA * LDP + kb];
            bf1[e] = Bs[buf][bnB * LDP + kb];
        }
        acc0 = __builtin_amdgcn_wmma_f32_16x16x32_f16(
                   false, af, false, bf0, (short)0, acc0, false, false);
        acc1 = __builtin_amdgcn_wmma_f32_16x16x32_f16(
                   false, af, false, bf1, (short)0, acc1, false, false);
    };

    // prologue: chunk 0
    bAsync(0, 0);
    gloadA(0);
    sstoreA(0);
    asm volatile("s_wait_asynccnt 0x0" ::: "memory");
    __syncthreads();

    for (int kc = 0; kc < nk; ++kc) {
        const int  cur  = kc & 1;
        const bool more = (kc + 1) < nk;
        if (more) {
            bAsync(kc + 1, cur ^ 1);   // async B copy in flight...
            gloadA(kc + 1);            // ...A loads in flight...
        }
        compute(cur);                  // ...both overlap the WMMAs
        if (more) sstoreA(cur ^ 1);
        asm volatile("s_wait_asynccnt 0x0" ::: "memory");
        __syncthreads();
    }

    // Epilogue: C/D layout -> lane lr = N col, VGPR v -> M = v + hi*8
    const int gnA = n0 + wn * 16 + lr;
    const int gnB = gnA + 32;
    const float bvA = (gnA < N && bias) ? bias[gnA] : 0.0f;
    const float bvB = (gnB < N && bias) ? bias[gnB] : 0.0f;
#pragma unroll
    for (int v = 0; v < 8; ++v) {
        const int gm = m0 + wm * 16 + v + hi * 8;
        float r0 = acc0[v] + bvA;
        float r1 = acc1[v] + bvB;
        if (MASKED) {
            if (rowmask[gm] == 0) { r0 = 0.0f; r1 = 0.0f; }
        }
        if (gnA < N) C[(long)gm * ldc + gnA] = r0;
        if (gnB < N) C[(long)gm * ldc + gnB] = r1;
    }
}

// ---------------------------------------------------------------------------
// Small helper kernels
// ---------------------------------------------------------------------------
__global__ void cvt_f32_f16_kernel(const float* __restrict__ src,
                                   _Float16* __restrict__ dst, int n)
{
    int i = blockIdx.x * blockDim.x + threadIdx.x;
    if (i < n) dst[i] = (_Float16)src[i];
}

// Fuse Wih (2048x768) and Whh (2048x512) into Wcat (2048x1280) f16
__global__ void build_wcat_kernel(const float* __restrict__ Wih,
                                  const float* __restrict__ Whh,
                                  _Float16* __restrict__ Wcat)
{
    long i = (long)blockIdx.x * blockDim.x + threadIdx.x;
    long total = (long)(4 * Dn) * (Mn + En + Dn);
    if (i >= total) return;
    int row = (int)(i / (Mn + En + Dn));
    int k   = (int)(i % (Mn + En + Dn));
    float v = (k < Mn + En) ? Wih[(long)row * (Mn + En) + k]
                            : Whh[(long)row * Dn + (k - (Mn + En))];
    Wcat[i] = (_Float16)v;
}

__global__ void build_bcat_kernel(const float* __restrict__ bih,
                                  const float* __restrict__ bhh,
                                  float* __restrict__ bcat)
{
    int i = blockIdx.x * blockDim.x + threadIdx.x;
    if (i < 4 * Dn) bcat[i] = bih[i] + bhh[i];
}

// mean over P: one block per batch, E threads
__global__ __launch_bounds__(En)
void mean_kernel(const float* __restrict__ enc, float* __restrict__ mean)
{
    int b = blockIdx.x, e = threadIdx.x;
    float s = 0.f;
    for (int p = 0; p < Pn; ++p)
        s += enc[((long)b * Pn + p) * En + e];
    mean[b * En + e] = s * (1.0f / Pn);
}

// Attention: scores -> softmax -> alphas (masked) -> ctx. One block per batch.
__global__ __launch_bounds__(256)
void attention_kernel(const float* __restrict__ enc,
                      const float* __restrict__ att1,
                      const float* __restrict__ att2,
                      const float* __restrict__ wfv,
                      const float* __restrict__ bfp,
                      const int* __restrict__ lengths, int t,
                      float* __restrict__ ctx,
                      float* __restrict__ alphas)
{
    __shared__ float a2s[An];
    __shared__ float sc[Pn + 4];
    __shared__ float sred[256];
    __shared__ float smax, ssum;

    const int b = blockIdx.x, tid = threadIdx.x;
    const int w = tid >> 5, lane = tid & 31;

    a2s[tid] = att2[b * An + tid];   // An == 256 == blockDim
    __syncthreads();

    const float bfs = bfp[0];
    // wave-parallel score dots: score[p] = relu(att1[b,p,:]+att2[b,:]) . wf + bf
    for (int p = w; p < Pn; p += 8) {
        float s = 0.f;
        const float* row = att1 + ((long)b * Pn + p) * An;
        for (int a = lane; a < An; a += 32) {
            float v = row[a] + a2s[a];
            v = fmaxf(v, 0.0f);
            s += v * wfv[a];
        }
#pragma unroll
        for (int off = 16; off > 0; off >>= 1)
            s += __shfl_down(s, off, 32);
        if (lane == 0) sc[p] = s + bfs;
    }
    __syncthreads();

    // softmax over P
    sred[tid] = (tid < Pn) ? sc[tid] : -1e30f;
    __syncthreads();
    for (int s = 128; s > 0; s >>= 1) {
        if (tid < s) sred[tid] = fmaxf(sred[tid], sred[tid + s]);
        __syncthreads();
    }
    if (tid == 0) smax = sred[0];
    __syncthreads();
    float ev = (tid < Pn) ? __expf(sc[tid] - smax) : 0.0f;
    sred[tid] = ev;
    __syncthreads();
    for (int s = 128; s > 0; s >>= 1) {
        if (tid < s) sred[tid] += sred[tid + s];
        __syncthreads();
    }
    if (tid == 0) ssum = sred[0];
    __syncthreads();
    if (tid < Pn) sc[tid] = ev / ssum;
    __syncthreads();

    const int active = (t < lengths[b] - 1) ? 1 : 0;
    if (tid < Pn)
        alphas[((long)b * Tn + t) * Pn + tid] = active ? sc[tid] : 0.0f;

    // ctx[b, e] = sum_p alpha[p] * enc[b, p, e]   (e = tid, coalesced)
    float acc = 0.f;
    for (int p = 0; p < Pn; ++p)
        acc += sc[p] * enc[((long)b * Pn + p) * En + tid];
    ctx[b * En + tid] = acc;
}

__global__ void gate_apply_kernel(const float* __restrict__ gatebuf,
                                  float* __restrict__ ctx)
{
    int i = blockIdx.x * blockDim.x + threadIdx.x;
    if (i < Bn * En) {
        float g = 1.0f / (1.0f + __expf(-gatebuf[i]));
        ctx[i] = g * ctx[i];
    }
}

// xh = [emb_t (512) | ctx (256) | h (512)]  -> (B x 1280) f32
__global__ void build_xh_kernel(const float* __restrict__ embW,
                                const int* __restrict__ captions, int t,
                                const float* __restrict__ ctx,
                                const float* __restrict__ h,
                                float* __restrict__ xh)
{
    int i = blockIdx.x * blockDim.x + threadIdx.x;
    const int KX = Mn + En + Dn;  // 1280
    if (i >= Bn * KX) return;
    int b = i / KX, k = i % KX;
    float v;
    if (k < Mn) {
        int tok = captions[b * Ln + t];
        v = embW[(long)tok * Mn + k];
    } else if (k < Mn + En) {
        v = ctx[b * En + (k - Mn)];
    } else {
        v = h[b * Dn + (k - Mn - En)];
    }
    xh[i] = v;
}

// LSTM pointwise: update h,c (masked), emit h_new and per-row active flags
__global__ void lstm_pointwise_kernel(const float* __restrict__ gates,
                                      const int* __restrict__ lengths, int t,
                                      float* __restrict__ h,
                                      float* __restrict__ c,
                                      float* __restrict__ hnew,
                                      int* __restrict__ active)
{
    int i = blockIdx.x * blockDim.x + threadIdx.x;
    if (i >= Bn * Dn) return;
    int b = i / Dn, d = i % Dn;
    const float* g = gates + (long)b * (4 * Dn);
    float ig = 1.0f / (1.0f + __expf(-g[d]));
    float fg = 1.0f / (1.0f + __expf(-g[Dn + d]));
    float gg = tanhf(g[2 * Dn + d]);
    float og = 1.0f / (1.0f + __expf(-g[3 * Dn + d]));
    float cn = fg * c[i] + ig * gg;
    float hn = og * tanhf(cn);
    int act = (t < lengths[b] - 1) ? 1 : 0;
    if (act) { h[i] = hn; c[i] = cn; }
    hnew[i] = hn;
    if (d == 0) active[b] = act;
}

// ---------------------------------------------------------------------------
// Host driver
// ---------------------------------------------------------------------------
static inline size_t alignup(size_t x) { return (x + 255) & ~(size_t)255; }

extern "C" void kernel_launch(void* const* d_in, const int* in_sizes, int n_in,
                              void* d_out, int out_size, void* d_ws, size_t ws_size,
                              hipStream_t stream)
{
    (void)in_sizes; (void)n_in; (void)out_size; (void)ws_size;

    const float* enc   = (const float*)d_in[0];
    const int*   caps  = (const int*)  d_in[1];
    const int*   lens  = (const int*)  d_in[2];
    const float* embW  = (const float*)d_in[3];
    const float* We    = (const float*)d_in[4];
    const float* be    = (const float*)d_in[5];
    const float* Wd    = (const float*)d_in[6];
    const float* bd    = (const float*)d_in[7];
    const float* wfv   = (const float*)d_in[8];
    const float* bfp   = (const float*)d_in[9];
    const float* Wih   = (const float*)d_in[10];
    const float* bih   = (const float*)d_in[11];
    const float* Whh   = (const float*)d_in[12];
    const float* bhh   = (const float*)d_in[13];
    const float* Wh0   = (const float*)d_in[14];
    const float* bh0   = (const float*)d_in[15];
    const float* Wc0   = (const float*)d_in[16];
    const float* bc0   = (const float*)d_in[17];
    const float* Wfb   = (const float*)d_in[18];
    const float* bfb   = (const float*)d_in[19];
    const float* Wfc   = (const float*)d_in[20];
    const float* bfc   = (const float*)d_in[21];

    float* preds  = (float*)d_out;                       // (B, T, V)
    float* alphas = preds + (size_t)Bn * Tn * Vn;        // (B, T, P)

    // Workspace layout
    char* ws = (char*)d_ws;
    size_t off = 0;
    auto take = [&](size_t bytes) { char* p = ws + off; off += alignup(bytes); return p; };

    const int KX = Mn + En + Dn;  // 1280
    _Float16* We16   = (_Float16*)take((size_t)An * En * 2);
    _Float16* Wd16   = (_Float16*)take((size_t)An * Dn * 2);
    _Float16* Wh016  = (_Float16*)take((size_t)Dn * En * 2);
    _Float16* Wc016  = (_Float16*)take((size_t)Dn * En * 2);
    _Float16* Wfb16  = (_Float16*)take((size_t)En * Dn * 2);
    _Float16* Wcat16 = (_Float16*)take((size_t)(4 * Dn) * KX * 2);
    _Float16* Wfc16  = (_Float16*)take((size_t)Vn * Dn * 2);
    float* bcat   = (float*)take((size_t)(4 * Dn) * 4);
    float* meanb  = (float*)take((size_t)Bn * En * 4);
    float* h      = (float*)take((size_t)Bn * Dn * 4);
    float* c      = (float*)take((size_t)Bn * Dn * 4);
    float* hnew   = (float*)take((size_t)Bn * Dn * 4);
    float* att1   = (float*)take((size_t)Bn * Pn * An * 4);
    float* att2   = (float*)take((size_t)Bn * An * 4);
    float* ctx    = (float*)take((size_t)Bn * En * 4);
    float* gatebf = (float*)take((size_t)Bn * En * 4);
    float* xh     = (float*)take((size_t)Bn * KX * 4);
    float* gates  = (float*)take((size_t)Bn * (4 * Dn) * 4);
    int*   active = (int*)  take((size_t)Bn * 4);

    auto cvt = [&](const float* s, _Float16* d, long n) {
        cvt_f32_f16_kernel<<<(int)((n + 255) / 256), 256, 0, stream>>>(s, d, (int)n);
    };

    // --- One-time setup: f16 weight conversions, fused Wih|Whh, mean, h0/c0, att1
    cvt(We,  We16,  (long)An * En);
    cvt(Wd,  Wd16,  (long)An * Dn);
    cvt(Wh0, Wh016, (long)Dn * En);
    cvt(Wc0, Wc016, (long)Dn * En);
    cvt(Wfb, Wfb16, (long)En * Dn);
    cvt(Wfc, Wfc16, (long)Vn * Dn);
    {
        long n = (long)(4 * Dn) * KX;
        build_wcat_kernel<<<(int)((n + 255) / 256), 256, 0, stream>>>(Wih, Whh, Wcat16);
        build_bcat_kernel<<<(4 * Dn + 255) / 256, 256, 0, stream>>>(bih, bhh, bcat);
    }
    mean_kernel<<<Bn, En, 0, stream>>>(enc, meanb);

    // M must be a multiple of 64 (128 and 25088 both are); K a multiple of 32.
    auto gemm = [&](const float* A, const _Float16* Bw, const float* bias,
                    float* C, long ldc, int M, int N, int K, const int* mask) {
        dim3 grid((N + 63) / 64, M / 64);
        if (mask)
            gemm_wmma_f16<1><<<grid, 256, 0, stream>>>(A, Bw, bias, C, ldc, N, K, mask);
        else
            gemm_wmma_f16<0><<<grid, 256, 0, stream>>>(A, Bw, bias, C, ldc, N, K, nullptr);
    };

    // h0 = mean @ Wh0^T + bh0 ; c0 = mean @ Wc0^T + bc0
    gemm(meanb, Wh016, bh0, h, Dn, Bn, Dn, En, nullptr);
    gemm(meanb, Wc016, bc0, c, Dn, Bn, Dn, En, nullptr);
    // att1 = enc_flat @ We^T + be   (25088 x 256 x 256)
    gemm(enc, We16, be, att1, An, Bn * Pn, An, En, nullptr);

    // --- Decode loop
    for (int t = 0; t < Tn; ++t) {
        // att2 = h @ Wd^T + bd
        gemm(h, Wd16, bd, att2, An, Bn, An, Dn, nullptr);
        // attention: softmax + alphas + ctx
        attention_kernel<<<Bn, 256, 0, stream>>>(enc, att1, att2, wfv, bfp,
                                                 lens, t, ctx, alphas);
        // gate = sigmoid(h @ Wfb^T + bfb); ctx *= gate
        gemm(h, Wfb16, bfb, gatebf, En, Bn, En, Dn, nullptr);
        gate_apply_kernel<<<(Bn * En + 255) / 256, 256, 0, stream>>>(gatebf, ctx);
        // xh = [emb_t | ctx | h]; gates = xh @ [Wih|Whh]^T + (bih+bhh)
        build_xh_kernel<<<(Bn * KX + 255) / 256, 256, 0, stream>>>(embW, caps, t, ctx, h, xh);
        gemm(xh, Wcat16, bcat, gates, 4 * Dn, Bn, 4 * Dn, KX, nullptr);
        // LSTM pointwise: update (h,c) masked, emit hnew + active flags
        lstm_pointwise_kernel<<<(Bn * Dn + 255) / 256, 256, 0, stream>>>(
            gates, lens, t, h, c, hnew, active);
        // pred[:, t, :] = active ? hnew @ Wfc^T + bfc : 0   (128 x 10000 x 512)
        gemm(hnew, Wfc16, bfc, preds + (size_t)t * Vn, (long)Tn * Vn, Bn, Vn, Dn, active);
    }
}
